// SlotAttention_42838003810433
// MI455X (gfx1250) — matrix operand order repaired and checked by reading, not compiled, not run on hardware
//
#include <hip/hip_runtime.h>

// ---------------------------------------------------------------------------
// SlotAttention-style truth->fastsim message passing for MI455X (gfx1250).
// Dense GEMMs (key/value/query MLPs, GRU gates, update MLP) run on
// v_wmma_f32_16x16x32_bf16 with weights pre-packed (once) into the WMMA
// B-fragment layout as bf16, so the inner loop is: 2x ds_load_b128 (A) +
// 2x global_load_b128 (B, cache-resident) + 1x v_wmma.
// Edge phase: deterministic wave-per-segment online softmax (dst sorted).
// ---------------------------------------------------------------------------

typedef __bf16 bf16_t;
typedef bf16_t v16bf __attribute__((ext_vector_type(16)));
typedef bf16_t v8bf  __attribute__((ext_vector_type(8)));
typedef bf16_t v2bf  __attribute__((ext_vector_type(2)));
typedef float  v8f   __attribute__((ext_vector_type(8)));

__device__ __forceinline__ float wave_max32(float v) {
#pragma unroll
  for (int o = 16; o > 0; o >>= 1) v = fmaxf(v, __shfl_xor(v, o, 32));
  return v;
}
__device__ __forceinline__ float wave_sum32(float v) {
#pragma unroll
  for (int o = 16; o > 0; o >>= 1) v += __shfl_xor(v, o, 32);
  return v;
}
__device__ __forceinline__ float sigmoidf_(float x) {
  return 1.0f / (1.0f + __expf(-x));
}
__device__ __forceinline__ float tanhf_(float x) {
  return 2.0f / (1.0f + __expf(-2.0f * x)) - 1.0f;
}

// ---------------------------------------------------------------------------
// Weight pre-pack: W f32 [Nout][K] row-major -> bf16 B-fragment tiles.
// Tile (ntile, ktile) covers cols [16*ntile,+16) x K [32*ktile,+32).
// Within a tile: lane L holds col N = 16*ntile + (L&15), K-chunks
// 32*ktile + [h*8,+8) and [16+h*8,+8), h=L>>4 -- 16 bf16 = 32 bytes per lane.
// tileIdx = ntile * (k_pad/32) + ktile ; out-of-range entries are 0.
// ---------------------------------------------------------------------------
__global__ __launch_bounds__(128) void prepack_kernel(
    const float* __restrict__ W, int Nout, int K, int k_pad, int n_pad,
    bf16_t* __restrict__ dst) {
  const int lane = threadIdx.x & 31;
  const int tile = blockIdx.x * (blockDim.x >> 5) + (threadIdx.x >> 5);
  const int ktiles = k_pad >> 5;
  const int ntile = tile / ktiles, ktile = tile - ntile * ktiles;
  if (ntile >= (n_pad >> 4)) return;
  const int n = ntile * 16 + (lane & 15);
  const int half = lane >> 4;
  bf16_t* out = dst + (size_t)tile * 512 + lane * 16;
#pragma unroll
  for (int j = 0; j < 16; ++j) {
    const int k = ktile * 32 + half * 8 + (j < 8 ? j : 8 + j);
    const float v = (n < Nout && k < K) ? W[(size_t)n * K + k] : 0.0f;
    out[j] = (bf16_t)v;
  }
}

// ---------------------------------------------------------------------------
// One 16x16 output tile of D = A @ W^T + b for a wave.
//   A:  LDS bf16 row-major [16][k_pad], zero-padded (k_pad mult of 32)
//   Wp: prepacked bf16 fragments with the SAME k_pad
// A-frag (ISA 7.12.2 16-bit A 16x32): lane L holds row M=L&15, K-chunks
// [h*8,+8) and [16+h*8,+8), h=L>>4.
// Columns >= n_valid produce exactly 0 (zero-packed weights + zero bias).
// ---------------------------------------------------------------------------
__device__ __forceinline__ v8f gemm_tile(const bf16_t* __restrict__ Alds, int k_pad,
                                         const bf16_t* __restrict__ Wp,
                                         const float* __restrict__ bias,
                                         int nb, int n_valid, int lane) {
  const int m    = lane & 15;
  const int half = lane >> 4;
  const int ktiles = k_pad >> 5;
  v8f acc = {0.f, 0.f, 0.f, 0.f, 0.f, 0.f, 0.f, 0.f};
  const bf16_t* bp = Wp + ((size_t)(nb >> 4) * ktiles) * 512 + lane * 16;
  const bf16_t* ar = Alds + m * k_pad + half * 8;
  for (int kt = 0; kt < ktiles; ++kt) {
    const v8bf alo = *(const v8bf*)(ar + kt * 32);
    const v8bf ahi = *(const v8bf*)(ar + kt * 32 + 16);
    const v16bf a = __builtin_shufflevector(alo, ahi, 0, 1, 2, 3, 4, 5, 6, 7,
                                            8, 9, 10, 11, 12, 13, 14, 15);
    const v16bf b = *(const v16bf*)(bp + kt * 512);
    acc = __builtin_amdgcn_wmma_f32_16x16x32_bf16(false, a, false, b,
                                                  (short)0, acc, false, false);
  }
  const int ncol = nb + m;
  const float bv = (ncol < n_valid) ? bias[ncol] : 0.0f;
#pragma unroll
  for (int r = 0; r < 8; ++r) acc[r] += bv;
  return acc;
}

// Full layer: OutLds[16][n_pad] = act(A @ W^T + b), bf16 result to LDS.
template <bool RELU>
__device__ __forceinline__ void mlp_layer(const bf16_t* __restrict__ Alds, int k_pad,
                                          const bf16_t* __restrict__ Wp,
                                          const float* __restrict__ bias, int n_valid,
                                          bf16_t* __restrict__ OutLds, int n_pad,
                                          int lane) {
  const int m = lane & 15, half = lane >> 4;
  for (int nb = 0; nb < n_pad; nb += 16) {
    v8f acc = gemm_tile(Alds, k_pad, Wp, bias, nb, n_valid, lane);
    const int ncol = nb + m;
#pragma unroll
    for (int r = 0; r < 8; ++r) {
      float v = acc[r];
      if (RELU) v = fmaxf(v, 0.0f);
      OutLds[(r + half * 8) * n_pad + ncol] = (bf16_t)v;
    }
  }
}

// ---------------------------------------------------------------------------
// Kernel 1: truth-side key (N_T x 30 -> padded 32, f32) and value-message
// (N_T x 72 = values MLP 64 cols ++ truth_features0 8 cols, bf16).
// 4 waves/block, 16 rows/wave.
// ---------------------------------------------------------------------------
__global__ __launch_bounds__(128) void truth_mlp_kernel(
    const float* __restrict__ tf, const float* __restrict__ tf0,
    const bf16_t* kw0, const float* kb0, const bf16_t* kw1, const float* kb1,
    const bf16_t* kw2, const float* kb2,
    const bf16_t* vw0, const float* vb0, const bf16_t* vw1, const float* vb1,
    const bf16_t* vw2, const float* vb2,
    float* __restrict__ key_out, bf16_t* __restrict__ valmsg, int n_rows) {
  __shared__ alignas(32) bf16_t sIn[4][16 * 96];
  __shared__ alignas(32) bf16_t sH0[4][16 * 64];
  __shared__ alignas(32) bf16_t sH1[4][16 * 64];
  const int wave = threadIdx.x >> 5, lane = threadIdx.x & 31;
  const int rb = (blockIdx.x * 4 + wave) * 16;
  if (rb >= n_rows) return;
  bf16_t* In = sIn[wave];
  bf16_t* H0 = sH0[wave];
  bf16_t* H1 = sH1[wave];

  for (int idx = lane; idx < 16 * 96; idx += 32) {
    const int r = idx / 96, c = idx % 96;
    const size_t row = (size_t)(rb + r);
    float v = 0.f;
    if (c < 64)      v = tf[row * 64 + c];
    else if (c < 72) v = tf0[row * 8 + (c - 64)];
    In[idx] = (bf16_t)v;
  }

  const int m = lane & 15, half = lane >> 4;

  // key path: 72->64->64->30 (stored padded to 32, pad cols exactly 0)
  mlp_layer<true>(In, 96, kw0, kb0, 64, H0, 64, lane);
  mlp_layer<true>(H0, 64, kw1, kb1, 64, H1, 64, lane);
  for (int nb = 0; nb < 32; nb += 16) {
    v8f acc = gemm_tile(H1, 64, kw2, kb2, nb, 30, lane);
    const int ncol = nb + m;
#pragma unroll
    for (int r = 0; r < 8; ++r)
      key_out[(size_t)(rb + r + half * 8) * 32 + ncol] = acc[r];
  }

  // values path: 72->64->64->64 -> bf16 valmsg[:, 0:64]
  mlp_layer<true>(In, 96, vw0, vb0, 64, H0, 64, lane);
  mlp_layer<true>(H0, 64, vw1, vb1, 64, H1, 64, lane);
  for (int nb = 0; nb < 64; nb += 16) {
    v8f acc = gemm_tile(H1, 64, vw2, vb2, nb, 64, lane);
    const int ncol = nb + m;
#pragma unroll
    for (int r = 0; r < 8; ++r)
      valmsg[(size_t)(rb + r + half * 8) * 72 + ncol] = (bf16_t)acc[r];
  }
  // valmsg[:, 64:72] = truth_features0
  for (int idx = lane; idx < 16 * 8; idx += 32) {
    const int r = idx / 8, c = idx % 8;
    valmsg[(size_t)(rb + r) * 72 + 64 + c] = (bf16_t)tf0[(size_t)(rb + r) * 8 + c];
  }
}

// ---------------------------------------------------------------------------
// Kernel 2: fastsim-side query MLP: [feat(64) ++ global(64)] 128->64->64->30
// stored padded to 32 f32 (pad cols 0, so 32-dot == 30-dot).
// ---------------------------------------------------------------------------
__global__ __launch_bounds__(128) void query_mlp_kernel(
    const float* __restrict__ ff, const float* __restrict__ fg,
    const bf16_t* qw0, const float* qb0, const bf16_t* qw1, const float* qb1,
    const bf16_t* qw2, const float* qb2, float* __restrict__ qry_out, int n_rows) {
  __shared__ alignas(32) bf16_t sIn[4][16 * 128];
  __shared__ alignas(32) bf16_t sH0[4][16 * 64];
  __shared__ alignas(32) bf16_t sH1[4][16 * 64];
  const int wave = threadIdx.x >> 5, lane = threadIdx.x & 31;
  const int rb = (blockIdx.x * 4 + wave) * 16;
  if (rb >= n_rows) return;
  bf16_t* In = sIn[wave];
  bf16_t* H0 = sH0[wave];
  bf16_t* H1 = sH1[wave];

  for (int idx = lane; idx < 16 * 128; idx += 32) {
    const int r = idx / 128, c = idx % 128;
    const size_t row = (size_t)(rb + r);
    const float v = (c < 64) ? ff[row * 64 + c] : fg[row * 64 + (c - 64)];
    In[idx] = (bf16_t)v;
  }
  mlp_layer<true>(In, 128, qw0, qb0, 64, H0, 64, lane);
  mlp_layer<true>(H0, 64, qw1, qb1, 64, H1, 64, lane);
  const int m = lane & 15, half = lane >> 4;
  for (int nb = 0; nb < 32; nb += 16) {
    v8f acc = gemm_tile(H1, 64, qw2, qb2, nb, 30, lane);
    const int ncol = nb + m;
#pragma unroll
    for (int r = 0; r < 8; ++r)
      qry_out[(size_t)(rb + r + half * 8) * 32 + ncol] = acc[r];
  }
}

// ---------------------------------------------------------------------------
// Kernel 3: one wave per destination segment (dst is sorted). Online
// (flash-style) segment softmax + 72-wide weighted message sum, no atomics.
// Lane owns message columns {2l, 2l+1} (+ {64+2l, 65+2l} for lanes 0..3),
// so gathers are b32 bf16-pair loads.
// ---------------------------------------------------------------------------
__global__ __launch_bounds__(128) void attn_kernel(
    const int* __restrict__ src, const int* __restrict__ dst,
    const float* __restrict__ key_t, const float* __restrict__ qry,
    const bf16_t* __restrict__ valmsg, float* __restrict__ wsum,
    int* __restrict__ deg, int E, int n_f) {
  const int lane = threadIdx.x & 31;
  const int d = blockIdx.x * 4 + (threadIdx.x >> 5);
  if (d >= n_f) return;

  // segment bounds via binary search (wave-uniform)
  int lo = 0, hi = E;
  while (lo < hi) { const int mid = (lo + hi) >> 1; if (dst[mid] < d) lo = mid + 1; else hi = mid; }
  const int start = lo;
  hi = E;
  while (lo < hi) { const int mid = (lo + hi) >> 1; if (dst[mid] <= d) lo = mid + 1; else hi = mid; }
  const int end = lo;
  const int degree = end - start;
  if (lane == 0) deg[d] = degree;

  float a0 = 0.f, a1 = 0.f, c0 = 0.f, c1 = 0.f, s = 0.f, m = -1e30f;
  if (degree > 0) {
    const float q = qry[(size_t)d * 32 + lane];  // lane l holds q[l]; q[30..31]==0
    const float norm = 0.18257418583505536f;     // 1/sqrt(30)
    for (int base = start; base < end; base += 32) {
      const int cnt = (end - base < 32) ? (end - base) : 32;
      const int e   = base + ((lane < cnt) ? lane : 0);
      const int se  = src[e];
      const float4* kr = (const float4*)(key_t + (size_t)se * 32);
      float acc = 0.f;
#pragma unroll
      for (int kk = 0; kk < 8; ++kk) {
        const float4 kv = kr[kk];
        acc += kv.x * __shfl(q, kk * 4 + 0, 32);
        acc += kv.y * __shfl(q, kk * 4 + 1, 32);
        acc += kv.z * __shfl(q, kk * 4 + 2, 32);
        acc += kv.w * __shfl(q, kk * 4 + 3, 32);
      }
      const float att = (lane < cnt) ? acc * norm : -1e30f;

      const float mc    = wave_max32(att);
      const float mnew  = fmaxf(m, mc);
      const float scale = __expf(m - mnew);
      s *= scale; a0 *= scale; a1 *= scale; c0 *= scale; c1 *= scale;
      const float w = (lane < cnt) ? __expf(att - mnew) : 0.f;
      s += wave_sum32(w);
      for (int j = 0; j < cnt; ++j) {
        const float wj = __shfl(w, j, 32);
        const int   sj = __shfl(se, j, 32);
        const bf16_t* vm = valmsg + (size_t)sj * 72;
        const v2bf p = *(const v2bf*)(vm + 2 * lane);
        a0 += wj * (float)p[0];
        a1 += wj * (float)p[1];
        if (lane < 4) {
          const v2bf t = *(const v2bf*)(vm + 64 + 2 * lane);
          c0 += wj * (float)t[0];
          c1 += wj * (float)t[1];
        }
      }
      m = mnew;
    }
  }
  const float inv = 1.0f / (s > 0.f ? s : 1.0f);
  float* out = wsum + (size_t)d * 72;
  out[2 * lane]     = a0 * inv;
  out[2 * lane + 1] = a1 * inv;
  if (lane < 4) {
    out[64 + 2 * lane]     = c0 * inv;
    out[64 + 2 * lane + 1] = c1 * inv;
  }
}

// ---------------------------------------------------------------------------
// Kernel 4: GRU cell (WMMA gates) + LayerNorm + 64->64->64 MLP + residual.
// ---------------------------------------------------------------------------
__global__ __launch_bounds__(128) void update_kernel(
    const float* __restrict__ wsumbuf, const float* __restrict__ fs,
    const bf16_t* wih, const float* bih, const bf16_t* whh, const float* bhh,
    const float* lng, const float* lnb,
    const bf16_t* mw0, const float* mb0, const bf16_t* mw1, const float* mb1,
    const int* __restrict__ deg, float* __restrict__ out, int n_rows) {
  __shared__ alignas(32) bf16_t sWs[4][16 * 96];
  __shared__ alignas(32) bf16_t sH[4][16 * 64];
  __shared__ alignas(32) float  sHp[4][16 * 64];
  __shared__ alignas(32) bf16_t sLn[4][16 * 64];
  __shared__ alignas(32) bf16_t sU0[4][16 * 64];
  const int wave = threadIdx.x >> 5, lane = threadIdx.x & 31;
  const int rb = (blockIdx.x * 4 + wave) * 16;
  if (rb >= n_rows) return;
  bf16_t* Ws = sWs[wave];
  bf16_t* H  = sH[wave];
  float*  Hp = sHp[wave];
  bf16_t* Ln = sLn[wave];
  bf16_t* U0 = sU0[wave];

  for (int idx = lane; idx < 16 * 96; idx += 32) {
    const int r = idx / 96, c = idx % 96;
    const float v = (c < 72) ? wsumbuf[(size_t)(rb + r) * 72 + c] : 0.f;
    Ws[idx] = (bf16_t)v;
  }
  for (int idx = lane; idx < 16 * 64; idx += 32) {
    const int r = idx / 64, c = idx % 64;
    H[idx] = (bf16_t)fs[(size_t)(rb + r) * 64 + c];
  }

  const int m16 = lane & 15, half = lane >> 4;
  v8f R[4], Z[4];
#pragma unroll
  for (int t = 0; t < 4; ++t) {  // reset gate: cols 0..63
    v8f gi = gemm_tile(Ws, 96, wih, bih, t * 16, 192, lane);
    v8f gh = gemm_tile(H, 64, whh, bhh, t * 16, 192, lane);
#pragma unroll
    for (int r = 0; r < 8; ++r) R[t][r] = sigmoidf_(gi[r] + gh[r]);
  }
#pragma unroll
  for (int t = 0; t < 4; ++t) {  // update gate: cols 64..127
    v8f gi = gemm_tile(Ws, 96, wih, bih, 64 + t * 16, 192, lane);
    v8f gh = gemm_tile(H, 64, whh, bhh, 64 + t * 16, 192, lane);
#pragma unroll
    for (int r = 0; r < 8; ++r) Z[t][r] = sigmoidf_(gi[r] + gh[r]);
  }
#pragma unroll
  for (int t = 0; t < 4; ++t) {  // candidate: cols 128..191, then h'
    v8f gi = gemm_tile(Ws, 96, wih, bih, 128 + t * 16, 192, lane);
    v8f gh = gemm_tile(H, 64, whh, bhh, 128 + t * 16, 192, lane);
    const int col = t * 16 + m16;
#pragma unroll
    for (int r = 0; r < 8; ++r) {
      const float nv = tanhf_(gi[r] + R[t][r] * gh[r]);
      const float hv = fs[(size_t)(rb + r + half * 8) * 64 + col];
      const float z  = Z[t][r];
      Hp[(r + half * 8) * 64 + col] = (1.f - z) * nv + z * hv;
    }
  }

  // LayerNorm rows (each lane owns cols lane and lane+32)
  for (int r = 0; r < 16; ++r) {
    const float x0 = Hp[r * 64 + lane];
    const float x1 = Hp[r * 64 + 32 + lane];
    const float mu = wave_sum32(x0 + x1) * (1.f / 64.f);
    const float d0 = x0 - mu, d1 = x1 - mu;
    const float var = wave_sum32(d0 * d0 + d1 * d1) * (1.f / 64.f);
    const float rs = rsqrtf(var + 1e-5f);
    Ln[r * 64 + lane]      = (bf16_t)(d0 * rs * lng[lane] + lnb[lane]);
    Ln[r * 64 + 32 + lane] = (bf16_t)(d1 * rs * lng[32 + lane] + lnb[32 + lane]);
  }

  // update MLP + masked residual
  mlp_layer<true>(Ln, 64, mw0, mb0, 64, U0, 64, lane);
  for (int nb = 0; nb < 64; nb += 16) {
    v8f acc = gemm_tile(U0, 64, mw1, mb1, nb, 64, lane);
    const int col = nb + m16;
#pragma unroll
    for (int r = 0; r < 8; ++r) {
      const int row = rb + r + half * 8;
      const float f = fs[(size_t)row * 64 + col];
      out[(size_t)row * 64 + col] = (deg[row] > 0) ? (f + acc[r]) : f;
    }
  }
}

// ---------------------------------------------------------------------------
// d_in layout (setup_inputs dict order, params flattened in insertion order):
//  0 truth_features  1 truth_features0  2 fastsim_features  3 fastsim_global
//  4 src  5 dst
//  6..11  key_{w0,b0,w1,b1,w2,b2}     12..17 query_{...}   18..23 values_{...}
//  24 gru_w_ih 25 gru_b_ih 26 gru_w_hh 27 gru_b_hh 28 ln_g 29 ln_b
//  30 mlp_w0 31 mlp_b0 32 mlp_w1 33 mlp_b1
// ---------------------------------------------------------------------------
extern "C" void kernel_launch(void* const* d_in, const int* in_sizes, int n_in,
                              void* d_out, int out_size, void* d_ws, size_t ws_size,
                              hipStream_t stream) {
  const float* tf  = (const float*)d_in[0];
  const float* tf0 = (const float*)d_in[1];
  const float* ff  = (const float*)d_in[2];
  const float* fg  = (const float*)d_in[3];
  const int*   src = (const int*)d_in[4];
  const int*   dst = (const int*)d_in[5];

  const int N_T = in_sizes[0] / 64;
  const int N_F = in_sizes[2] / 64;
  const int E   = in_sizes[4];

  char* ws = (char*)d_ws;
  size_t off = 0;
  auto carve = [&](size_t bytes) {
    void* p = ws + off;
    off = (off + bytes + 255) & ~(size_t)255;
    return p;
  };

  // --- prepacked weights (bf16 WMMA B-frag layout), ~150 KB total ---------
  struct PackDesc { int idx; int Nout, K, k_pad, n_pad; bf16_t* dst; };
  PackDesc pk[13] = {
      {6,  64, 72, 96, 64, nullptr},   // key_w0
      {8,  64, 64, 64, 64, nullptr},   // key_w1
      {10, 30, 64, 64, 32, nullptr},   // key_w2
      {12, 64, 128, 128, 64, nullptr}, // query_w0
      {14, 64, 64, 64, 64, nullptr},   // query_w1
      {16, 30, 64, 64, 32, nullptr},   // query_w2
      {18, 64, 72, 96, 64, nullptr},   // values_w0
      {20, 64, 64, 64, 64, nullptr},   // values_w1
      {22, 64, 64, 64, 64, nullptr},   // values_w2
      {24, 192, 72, 96, 192, nullptr}, // gru_w_ih
      {26, 192, 64, 64, 192, nullptr}, // gru_w_hh
      {30, 64, 64, 64, 64, nullptr},   // mlp_w0
      {32, 64, 64, 64, 64, nullptr},   // mlp_w1
  };
  for (int i = 0; i < 13; ++i) {
    const int tiles = (pk[i].n_pad >> 4) * (pk[i].k_pad >> 5);
    pk[i].dst = (bf16_t*)carve((size_t)tiles * 512 * sizeof(bf16_t));
  }

  // --- activations scratch ------------------------------------------------
  float*  key_t   = (float*)carve((size_t)N_T * 32 * sizeof(float));
  bf16_t* valmsg  = (bf16_t*)carve((size_t)N_T * 72 * sizeof(bf16_t));
  float*  qry     = (float*)carve((size_t)N_F * 32 * sizeof(float));
  float*  wsumbuf = (float*)carve((size_t)N_F * 72 * sizeof(float));
  int*    deg     = (int*)carve((size_t)N_F * sizeof(int));

  for (int i = 0; i < 13; ++i) {
    const int tiles = (pk[i].n_pad >> 4) * (pk[i].k_pad >> 5);
    prepack_kernel<<<dim3((tiles + 3) / 4), dim3(128), 0, stream>>>(
        (const float*)d_in[pk[i].idx], pk[i].Nout, pk[i].K, pk[i].k_pad,
        pk[i].n_pad, pk[i].dst);
  }

  const float* kb0 = (const float*)d_in[7];
  const float* kb1 = (const float*)d_in[9];
  const float* kb2 = (const float*)d_in[11];
  const float* qb0 = (const float*)d_in[13];
  const float* qb1 = (const float*)d_in[15];
  const float* qb2 = (const float*)d_in[17];
  const float* vb0 = (const float*)d_in[19];
  const float* vb1 = (const float*)d_in[21];
  const float* vb2 = (const float*)d_in[23];
  const float* bih = (const float*)d_in[25];
  const float* bhh = (const float*)d_in[27];
  const float* lng = (const float*)d_in[28];
  const float* lnb = (const float*)d_in[29];
  const float* mb0 = (const float*)d_in[31];
  const float* mb1 = (const float*)d_in[33];

  const dim3 blk(128);
  truth_mlp_kernel<<<dim3((N_T + 63) / 64), blk, 0, stream>>>(
      tf, tf0, pk[0].dst, kb0, pk[1].dst, kb1, pk[2].dst, kb2,
      pk[6].dst, vb0, pk[7].dst, vb1, pk[8].dst, vb2, key_t, valmsg, N_T);
  query_mlp_kernel<<<dim3((N_F + 63) / 64), blk, 0, stream>>>(
      ff, fg, pk[3].dst, qb0, pk[4].dst, qb1, pk[5].dst, qb2, qry, N_F);
  attn_kernel<<<dim3((N_F + 3) / 4), blk, 0, stream>>>(
      src, dst, key_t, qry, valmsg, wsumbuf, deg, E, N_F);
  update_kernel<<<dim3((N_F + 63) / 64), blk, 0, stream>>>(
      wsumbuf, ff, pk[9].dst, bih, pk[10].dst, bhh, lng, lnb,
      pk[11].dst, mb0, pk[12].dst, mb1, deg, (float*)d_out, N_F);
}
